// GlobalMQA_40939628265847
// MI455X (gfx1250) — matrix operand order repaired and checked by reading, not compiled, not run on hardware
//
#include <hip/hip_runtime.h>

// ---------------------------------------------------------------------------
// MQA attention block (RoPE + flash attention + projections) for gfx1250.
// All matmuls run through v_wmma_f32_16x16x32_bf16 (bf16 in, f32 accumulate).
// Attention K/V tiles stream through global_load_async_to_lds_b128 with LDS
// double-buffering (ASYNCcnt + s_wait_asynccnt), falling back to direct
// copies if the async builtins are unavailable.
// ---------------------------------------------------------------------------

typedef __bf16 bf16;
typedef __attribute__((ext_vector_type(4)))  bf16 v4bf;
typedef __attribute__((ext_vector_type(8)))  bf16 v8bf;
typedef __attribute__((ext_vector_type(16))) bf16 v16bf;
typedef __attribute__((ext_vector_type(8)))  float v8f;

#define BATCH  2
#define SEQ    2048
#define DMODEL 1024
#define NHEAD  8
#define DHEAD  128
#define NCOLS  1280   // 1024 Q cols | 128 K cols | 128 V cols
#define SM_SCALE 0.08838834764831845f  // 1/sqrt(128)

#if __has_builtin(__builtin_amdgcn_global_load_async_to_lds_b128) && \
    __has_builtin(__builtin_amdgcn_s_wait_asynccnt)
#define ASYNC_LDS 1
#else
#define ASYNC_LDS 0
#endif

static __device__ __forceinline__ v8f wmma_bf16(v16bf a, v16bf b, v8f c) {
  return __builtin_amdgcn_wmma_f32_16x16x32_bf16(false, a, false, b, (short)0, c,
                                                 false, false);
}
static __device__ __forceinline__ v8bf ld8(const bf16* p) { return *(const v8bf*)p; }
static __device__ __forceinline__ v16bf cat8(v8bf lo, v8bf hi) {
  return __builtin_shufflevector(lo, hi, 0,1,2,3,4,5,6,7,8,9,10,11,12,13,14,15);
}
// B-fragment load, table order (lanes 0-15: K0..7,16..23; lanes 16-31: K8..15,24..31)
static __device__ __forceinline__ v16bf ld16(const bf16* p) { return cat8(ld8(p), ld8(p + 16)); }
static __device__ __forceinline__ v8f zero8() { v8f z = {0.f,0.f,0.f,0.f,0.f,0.f,0.f,0.f}; return z; }

// 16B global -> LDS copy, async on CDNA5 (tracked by ASYNCcnt)
static __device__ __forceinline__ void cp16_g2l(const bf16* g, bf16* l) {
#if ASYNC_LDS
  typedef int v4i_t __attribute__((vector_size(16)));
  typedef __attribute__((address_space(1))) v4i_t* gp_t;
  typedef __attribute__((address_space(3))) v4i_t* lp_t;
  __attribute__((address_space(1))) void* gv = (__attribute__((address_space(1))) void*)g;
  __attribute__((address_space(3))) void* lv = (__attribute__((address_space(3))) void*)l;
  __builtin_amdgcn_global_load_async_to_lds_b128((gp_t)gv, (lp_t)lv, 0, 0);
#else
  *(uint4*)l = *(const uint4*)g;
#endif
}

// ---------------------------------------------------------------------------
// Prep kernels: x -> bf16; weights -> [col][k] bf16 (transposed for B-frags)
// ---------------------------------------------------------------------------
__global__ __launch_bounds__(256) void k_cvt_x(const float* __restrict__ x,
                                               bf16* __restrict__ xb) {
  const int i = blockIdx.x * 256 + threadIdx.x;           // group of 4 floats
  const float4 v = ((const float4*)x)[i];
  v4bf o; o[0] = (bf16)v.x; o[1] = (bf16)v.y; o[2] = (bf16)v.z; o[3] = (bf16)v.w;
  ((v4bf*)xb)[i] = o;
}

__global__ __launch_bounds__(256) void k_prep_w(const float* __restrict__ Wq,
                                                const float* __restrict__ Wk,
                                                const float* __restrict__ Wv,
                                                const float* __restrict__ Wo,
                                                bf16* __restrict__ Wt,
                                                bf16* __restrict__ Wot) {
  const int i = blockIdx.x * 256 + threadIdx.x;
  if (i < NCOLS * DMODEL) {              // Wt[c][k] = W*[k][c]
    const int c = i >> 10, k = i & 1023;
    float v;
    if (c < 1024)       v = Wq[k * 1024 + c];
    else if (c < 1152)  v = Wk[k * 128 + (c - 1024)];
    else                v = Wv[k * 128 + (c - 1152)];
    Wt[i] = (bf16)v;
  } else {
    const int j = i - NCOLS * DMODEL;    // Wot[c][k] = Wo[k][c]
    const int c = j >> 10, k = j & 1023;
    Wot[j] = (bf16)Wo[k * 1024 + c];
  }
}

// ---------------------------------------------------------------------------
// Kernel 1: fused QKV projection + RoPE epilogue.
// Wave tile: 16 rows x 128 cols (one full head slice -> RoPE pairs d and d+64
// live in C-frags t and t+4, same lane/element => pure in-register rotation).
// Outputs: Q bf16 [b][h][n][128], K bf16 [b][n][128], V bf16 transposed [b][128][n].
// ---------------------------------------------------------------------------
__global__ __launch_bounds__(256) void k_qkv_rope(const bf16* __restrict__ xb,
                                                  const bf16* __restrict__ Wt,
                                                  bf16* __restrict__ qbf,
                                                  bf16* __restrict__ kbf,
                                                  bf16* __restrict__ vtr) {
  const int lane = threadIdx.x & 31;
  const int wave = (blockIdx.x * blockDim.x + threadIdx.x) >> 5;   // 0..2559
  const int l16 = lane & 15, hf = lane >> 4;
  const int mt = wave / 10;              // 256 row tiles
  const int nb = wave - mt * 10;         // 10 col blocks of 128
  const int m0 = mt * 16;
  const int colbase = nb * 128;

  v8f acc[8];
#pragma unroll
  for (int t = 0; t < 8; ++t) acc[t] = zero8();

  const bf16* arow = xb + (size_t)(m0 + l16) * DMODEL;
  const bf16* bcol = Wt + (size_t)(colbase + l16) * DMODEL;
  for (int kk = 0; kk < DMODEL; kk += 32) {
    // A-frag: lo = K(kk + hf*8 ..+8), hi = K(kk+16+hf*8 ..+8)
    const v16bf a = cat8(ld8(arow + kk + hf * 8), ld8(arow + kk + 16 + hf * 8));
    v16bf bfr[8];
#pragma unroll
    for (int t = 0; t < 8; ++t)
      bfr[t] = ld16(bcol + (size_t)t * 16 * DMODEL + kk + hf * 8);
    if (kk + 32 < DMODEL) __builtin_prefetch(arow + kk + 32, 0, 0);
#pragma unroll
    for (int t = 0; t < 8; ++t) acc[t] = wmma_bf16(a, bfr[t], acc[t]);
  }

  if (colbase < 1152) {  // RoPE for Q blocks (0..896) and K block (1024)
#pragma unroll
    for (int t = 0; t < 4; ++t) {
      const int d1 = t * 16 + l16;                         // 0..63
      const float freq = __expf(-(float)d1 * (9.2103403719761836f / 64.0f));
#pragma unroll
      for (int r = 0; r < 8; ++r) {
        const int rowg = m0 + r + 8 * hf;
        const int p = rowg & (SEQ - 1);                    // token position
        float s, c;
        __sincosf((float)p * freq, &s, &c);
        const float x1 = acc[t][r], x2 = acc[t + 4][r];
        acc[t][r]     = x1 * c - x2 * s;
        acc[t + 4][r] = x2 * c + x1 * s;
      }
    }
  }

#pragma unroll
  for (int t = 0; t < 8; ++t) {
    const int cidx = colbase + t * 16 + l16;
#pragma unroll
    for (int r = 0; r < 8; ++r) {
      const int rowg = m0 + r + 8 * hf;
      const int bb = rowg >> 11;
      const int p = rowg & (SEQ - 1);
      const bf16 val = (bf16)acc[t][r];
      if (cidx < 1024) {
        const int hh = cidx >> 7, dd = cidx & 127;
        qbf[(((size_t)bb * NHEAD + hh) * SEQ + p) * DHEAD + dd] = val;
      } else if (cidx < 1152) {
        kbf[((size_t)bb * SEQ + p) * DHEAD + (cidx - 1024)] = val;
      } else {
        vtr[((size_t)bb * DHEAD + (cidx - 1152)) * SEQ + p] = val;  // V transposed
      }
    }
  }
}

// ---------------------------------------------------------------------------
// Kernel 2: flash-attention MQA with async double-buffered K/V staging.
// Block = 8 waves x 32 lanes, 128 queries of one (b,h); 32-key tiles of the
// shared K/V stream staged in LDS. S^T = K*Q^T so each lane owns one query
// column; the P tile in C-layout is exactly the A-fragment layout needed for
// P*V -> in-register softmax->PV handoff.
// ---------------------------------------------------------------------------
__global__ __launch_bounds__(256) void k_attn(const bf16* __restrict__ qbf,
                                              const bf16* __restrict__ kbf,
                                              const bf16* __restrict__ vtr,
                                              bf16* __restrict__ aout) {
  __shared__ __align__(16) bf16 Kt[2][32 * DHEAD];   // [key][d]  2 x 8 KB
  __shared__ __align__(16) bf16 Vt[2][DHEAD * 32];   // [d][key]  2 x 8 KB

  const int tid = threadIdx.x;
  const int lane = tid & 31, w = tid >> 5;
  const int l16 = lane & 15, hf = lane >> 4;
  const int bh = blockIdx.x >> 4;
  const int bb = bh >> 3, hh = bh & 7;
  const int q0 = (blockIdx.x & 15) * 128 + w * 16;

  const bf16* kbase = kbf + (size_t)bb * SEQ * DHEAD;
  const bf16* vbase = vtr + (size_t)bb * DHEAD * SEQ;
  const int dR = tid >> 1, part = tid & 1;   // V-tile staging coords

  // stage tile `it` into buffer `bufi` (4 x 16B per thread, async on CDNA5)
  auto stage = [&](int it, int bufi) {
    const int j0 = it * 32;
    cp16_g2l(kbase + (size_t)j0 * DHEAD + tid * 8,        &Kt[bufi][tid * 8]);
    cp16_g2l(kbase + (size_t)j0 * DHEAD + 2048 + tid * 8, &Kt[bufi][2048 + tid * 8]);
    cp16_g2l(vbase + (size_t)dR * SEQ + j0 + part * 16,     &Vt[bufi][dR * 32 + part * 16]);
    cp16_g2l(vbase + (size_t)dR * SEQ + j0 + part * 16 + 8, &Vt[bufi][dR * 32 + part * 16 + 8]);
  };

  // Q B-frags (Q^T: K-dim = d, N = 16 queries), resident for whole kernel
  v16bf qf[4];
  {
    const bf16* qrow = qbf + (((size_t)bb * NHEAD + hh) * SEQ + q0 + l16) * DHEAD;
#pragma unroll
    for (int dd = 0; dd < 4; ++dd) qf[dd] = ld16(qrow + dd * 32 + hf * 8);
  }

  v8f o[8];
#pragma unroll
  for (int t = 0; t < 8; ++t) o[t] = zero8();
  float mrun = -1e30f, lsum = 0.0f;

  stage(0, 0);                                    // prologue: tile 0 -> buf 0

  for (int it = 0; it < SEQ / 32; ++it) {
    const int buf = it & 1;
#if ASYNC_LDS
    __builtin_amdgcn_s_wait_asynccnt(0);          // my tile-`it` loads landed
#endif
    __syncthreads();                              // everyone's landed; buf^1 free
    if (it + 1 < SEQ / 32) stage(it + 1, buf ^ 1);  // overlap next tile w/ compute

    // S^T tiles (keys x queries): A = K rows from LDS, B = Q^T in regs
    v8f st0 = zero8(), st1 = zero8();
#pragma unroll
    for (int dd = 0; dd < 4; ++dd) {
      const bf16* kr0 = &Kt[buf][l16 * DHEAD + dd * 32];
      const bf16* kr1 = &Kt[buf][(16 + l16) * DHEAD + dd * 32];
      const v16bf a0 = cat8(ld8(kr0 + hf * 8), ld8(kr0 + 16 + hf * 8));
      const v16bf a1 = cat8(ld8(kr1 + hf * 8), ld8(kr1 + 16 + hf * 8));
      st0 = wmma_bf16(a0, qf[dd], st0);
      st1 = wmma_bf16(a1, qf[dd], st1);
    }

    // Online softmax: lane owns query l16; partner half via one shfl_xor(16)
    float vm = -1e30f;
#pragma unroll
    for (int r = 0; r < 8; ++r) {
      st0[r] *= SM_SCALE; st1[r] *= SM_SCALE;
      vm = fmaxf(vm, fmaxf(st0[r], st1[r]));
    }
    vm = fmaxf(vm, __shfl_xor(vm, 16, 32));
    const float mnew = fmaxf(mrun, vm);
    const float alpha = __expf(mrun - mnew);
    mrun = mnew;

    v8bf p0, p1;
    float rsum = 0.0f;
#pragma unroll
    for (int r = 0; r < 8; ++r) {
      const float e0 = __expf(st0[r] - mnew);
      const float e1 = __expf(st1[r] - mnew);
      rsum += e0 + e1;
      p0[r] = (bf16)e0; p1[r] = (bf16)e1;
    }
    rsum += __shfl_xor(rsum, 16, 32);
    lsum = lsum * alpha + rsum;

#pragma unroll
    for (int t = 0; t < 8; ++t)
#pragma unroll
      for (int r = 0; r < 8; ++r) o[t][r] *= alpha;

    // P in C-layout == A-frag layout for P*V (same lane, same element order)
    const v16bf pa = cat8(p0, p1);
#pragma unroll
    for (int dt = 0; dt < 8; ++dt) {
      const v16bf vb = ld16(&Vt[buf][(dt * 16 + l16) * 32 + hf * 8]);
      o[dt] = wmma_bf16(pa, vb, o[dt]);
    }
  }

  // Normalize and emit [b][q][h*128+d] bf16 for the output GEMM
#pragma unroll
  for (int r = 0; r < 8; ++r) {
    const float ls = __shfl(lsum, r + 8 * hf, 32);
    const float inv = 1.0f / ls;
    const int qg = q0 + r + 8 * hf;
    bf16* orow = aout + ((size_t)bb * SEQ + qg) * (NHEAD * DHEAD) + hh * DHEAD;
#pragma unroll
    for (int dt = 0; dt < 8; ++dt) orow[dt * 16 + l16] = (bf16)(o[dt][r] * inv);
  }
}

// ---------------------------------------------------------------------------
// Kernel 3: output projection  out = attn @ Wo + bo   (f32 result)
// ---------------------------------------------------------------------------
__global__ __launch_bounds__(256) void k_out(const bf16* __restrict__ aout,
                                             const bf16* __restrict__ Wot,
                                             const float* __restrict__ bo,
                                             float* __restrict__ out) {
  const int lane = threadIdx.x & 31;
  const int wave = (blockIdx.x * blockDim.x + threadIdx.x) >> 5;   // 0..2047
  const int l16 = lane & 15, hf = lane >> 4;
  const int mt = wave >> 3, nb = wave & 7;
  const int m0 = mt * 16, colbase = nb * 128;

  v8f acc[8];
#pragma unroll
  for (int t = 0; t < 8; ++t) acc[t] = zero8();

  const bf16* arow = aout + (size_t)(m0 + l16) * DMODEL;
  const bf16* bcol = Wot + (size_t)(colbase + l16) * DMODEL;
  for (int kk = 0; kk < DMODEL; kk += 32) {
    const v16bf a = cat8(ld8(arow + kk + hf * 8), ld8(arow + kk + 16 + hf * 8));
    v16bf bfr[8];
#pragma unroll
    for (int t = 0; t < 8; ++t)
      bfr[t] = ld16(bcol + (size_t)t * 16 * DMODEL + kk + hf * 8);
    if (kk + 32 < DMODEL) __builtin_prefetch(arow + kk + 32, 0, 0);
#pragma unroll
    for (int t = 0; t < 8; ++t) acc[t] = wmma_bf16(a, bfr[t], acc[t]);
  }

#pragma unroll
  for (int t = 0; t < 8; ++t) {
    const int col = colbase + t * 16 + l16;
    const float bias = bo[col];
#pragma unroll
    for (int r = 0; r < 8; ++r) {
      out[(size_t)(m0 + r + 8 * hf) * DMODEL + col] = acc[t][r] + bias;
    }
  }
}

// ---------------------------------------------------------------------------
extern "C" void kernel_launch(void* const* d_in, const int* in_sizes, int n_in,
                              void* d_out, int out_size, void* d_ws, size_t ws_size,
                              hipStream_t stream) {
  (void)in_sizes; (void)n_in; (void)out_size; (void)ws_size;
  const float* x  = (const float*)d_in[0];
  const float* Wq = (const float*)d_in[1];
  const float* Wk = (const float*)d_in[2];
  const float* Wv = (const float*)d_in[3];
  const float* Wo = (const float*)d_in[4];
  const float* bo = (const float*)d_in[5];
  float* out = (float*)d_out;

  char* p = (char*)d_ws;
  bf16* xb   = (bf16*)p; p += (size_t)4096 * 1024 * 2;              // x bf16
  bf16* Wt   = (bf16*)p; p += (size_t)NCOLS * 1024 * 2;             // [col][k] qkv
  bf16* Wot  = (bf16*)p; p += (size_t)1024 * 1024 * 2;              // [col][k] Wo
  bf16* qbf  = (bf16*)p; p += (size_t)BATCH * NHEAD * SEQ * DHEAD * 2;
  bf16* kbf  = (bf16*)p; p += (size_t)BATCH * SEQ * DHEAD * 2;
  bf16* vtr  = (bf16*)p; p += (size_t)BATCH * DHEAD * SEQ * 2;      // V transposed
  bf16* aout = (bf16*)p; p += (size_t)4096 * 1024 * 2;              // attn output

  k_cvt_x   <<<4096, 256, 0, stream>>>(x, xb);
  k_prep_w  <<<9216, 256, 0, stream>>>(Wq, Wk, Wv, Wo, Wt, Wot);
  k_qkv_rope<<<320,  256, 0, stream>>>(xb, Wt, qbf, kbf, vtr);
  k_attn    <<<256,  256, 0, stream>>>(qbf, kbf, vtr, aout);
  k_out     <<<256,  256, 0, stream>>>(aout, Wot, bo, out);
}
